// ModuleCorefProp2_24790551232667
// MI455X (gfx1250) — matrix operand order repaired and checked by reading, not compiled, not run on hardware
//
#include <hip/hip_runtime.h>
#include <hip/hip_bf16.h>

// ---- problem constants (B=1) ----
static constexpr int SK  = 512;    // pruned spans K
static constexpr int SD  = 512;    // span dim D
static constexpr int SH  = 64;     // pair hidden H
static constexpr int SN  = 16384;  // all spans N
static constexpr int SNB = 10;     // distance buckets
static constexpr int SDD = 64;     // dist emb dim
static constexpr float BIGF = 1e23f;

typedef __bf16 bf16_t;
typedef __bf16 v16bf __attribute__((ext_vector_type(16)));
typedef float  v8f   __attribute__((ext_vector_type(8)));

union Frag16 { v16bf v; uint4 q[2]; unsigned u[8]; };
union FragC  { v8f   v; float f[8]; };

__device__ inline bf16_t f2bf(float f) {
    union { float f; unsigned u; } x; x.f = f;
    unsigned r = (x.u + 0x7FFFu + ((x.u >> 16) & 1u)) >> 16;   // RNE
    union { unsigned short s; bf16_t b; } y; y.s = (unsigned short)r;
    return y.b;
}

__device__ inline unsigned pk_mul_bf16(unsigned a, unsigned b) {
    unsigned d;
    asm volatile("v_pk_mul_bf16 %0, %1, %2" : "=v"(d) : "v"(a), "v"(b));
    return d;
}
__device__ inline uint4 pk_mul4(uint4 a, uint4 b) {
    uint4 r;
    r.x = pk_mul_bf16(a.x, b.x); r.y = pk_mul_bf16(a.y, b.y);
    r.z = pk_mul_bf16(a.z, b.z); r.w = pk_mul_bf16(a.w, b.w);
    return r;
}

// ---------------- small prep kernels ----------------

// dst[c*rows + r] = bf16(src[r*cols + c])   (builds B^T operands)
__global__ void k_transpose_bf16(bf16_t* __restrict__ dst, const float* __restrict__ src,
                                 int rows, int cols) {
    int idx = blockIdx.x * blockDim.x + threadIdx.x;
    if (idx >= rows * cols) return;
    int r = idx / cols, c = idx - r * cols;
    dst[(size_t)c * rows + r] = f2bf(src[idx]);
}

__global__ void k_cvt_bf16(bf16_t* __restrict__ dst, const float* __restrict__ src, int n) {
    int idx = blockIdx.x * blockDim.x + threadIdx.x;
    if (idx < n) dst[idx] = f2bf(src[idx]);
}

// bucket_feat[nb][h] = dist_emb[nb] @ Wd[:,h] + bd[h]   (NB*H = 640 outputs)
__global__ void k_bucket_feat(const float* __restrict__ de, const float* __restrict__ Wd,
                              const float* __restrict__ bd, float* __restrict__ out) {
    int idx = blockIdx.x * blockDim.x + threadIdx.x;
    if (idx >= SNB * SH) return;
    int nb = idx / SH, h = idx - nb * SH;
    float s = bd[h];
    for (int dd = 0; dd < SDD; ++dd) s += de[nb * SDD + dd] * Wd[dd * SH + h];
    out[idx] = s;
}

// ---------------- generic bf16 WMMA GEMM:  C[M x N] = A[M x Kd] @ Bt[N x Kd]^T (+bias[N]) ----
__global__ __launch_bounds__(32)
void k_gemm_bt(const bf16_t* __restrict__ A, const bf16_t* __restrict__ Bt,
               const float* __restrict__ bias, float* __restrict__ C,
               int M, int Nn, int Kd) {
    const int lane = threadIdx.x & 31;
    const int hi   = lane >> 4;
    const int mn   = lane & 15;
    const int j0   = blockIdx.x * 16;
    const int i0   = blockIdx.y * 16;

    const uint4* ap = (const uint4*)(A  + (size_t)(i0 + mn) * Kd);
    const uint4* bp = (const uint4*)(Bt + (size_t)(j0 + mn) * Kd);

    FragC acc;
#pragma unroll
    for (int e = 0; e < 8; ++e) acc.f[e] = 0.0f;

    for (int kk = 0; kk < Kd / 32; ++kk) {
        Frag16 a, b;
        int c0 = kk * 4 + hi;          // A: elems kk*32 + hi*8  (lane halves hold K {0-7,16-23}/{8-15,24-31})
        a.q[0] = ap[c0];
        a.q[1] = ap[c0 + 2];           // +16 elements
        int bi = kk * 4 + hi;          // B: same K split as A ({0-7,16-23}/{8-15,24-31}) of column mn
        b.q[0] = bp[bi];
        b.q[1] = bp[bi + 2];
        acc.v = __builtin_amdgcn_wmma_f32_16x16x32_bf16(false, a.v, false, b.v,
                                                        (short)0, acc.v, false, false);
    }
#pragma unroll
    for (int e = 0; e < 8; ++e) {
        int m = hi * 8 + e;
        int gi = i0 + m, gj = j0 + mn;
        float v = acc.f[e];
        if (bias) v += bias[gj];
        C[(size_t)gi * Nn + gj] = v;
    }
}

// ---------------- fused pairwise scorer ----------------
// cs[i,j] = sum_h relu(left[i,h]+right[j,h]+ (u_i ⊙ Wp[:,h]) . u_j + bfeat[bucket(i,j),h]) * Wo[h]
//           + bo + svec[i] + svec[j],  diag zeroed.
__global__ __launch_bounds__(128)
void k_scorer(const bf16_t* __restrict__ u_bf,   // [SK][SD] bf16
              const bf16_t* __restrict__ wpT,    // [SH][SD] bf16 (Wp transposed)
              const float*  __restrict__ leftp,  // [SK][SH]
              const float*  __restrict__ rightp, // [SK][SH]
              const float*  __restrict__ bfeat,  // [SNB][SH]
              const float*  __restrict__ Wo,     // [SH]
              const float*  __restrict__ bo,     // [1]
              const float*  __restrict__ svec,   // [SK]
              const int*    __restrict__ span_begin, // [SK]
              float* __restrict__ cs) {          // [SK][SK]
    __shared__ __align__(16) bf16_t su_i[16][SD];
    __shared__ __align__(16) bf16_t su_j[16][SD];
    __shared__ __align__(16) float  sleft[16][SH];
    __shared__ __align__(16) float  sright[16][SH];
    __shared__ __align__(16) float  sbf[SNB][SH];
    __shared__ float swo[SH];
    __shared__ int   sbid[16][16];
    __shared__ float stile[256];

    const int j0 = blockIdx.x * 16;
    const int i0 = blockIdx.y * 16;
    const int t  = threadIdx.x;    // 0..127

    { // stage u tiles (16 rows x 512 bf16, contiguous)
        const uint4* si = (const uint4*)(u_bf + (size_t)i0 * SD);
        const uint4* sj = (const uint4*)(u_bf + (size_t)j0 * SD);
        uint4* di = (uint4*)&su_i[0][0];
        uint4* dj = (uint4*)&su_j[0][0];
        for (int v = t; v < 16 * SD / 8; v += 128) { di[v] = si[v]; dj[v] = sj[v]; }
    }
    for (int v = t; v < 16 * SH; v += 128) {
        ((float*)&sleft[0][0])[v]  = leftp [(size_t)i0 * SH + v];
        ((float*)&sright[0][0])[v] = rightp[(size_t)j0 * SH + v];
    }
    for (int v = t; v < SNB * SH; v += 128) ((float*)&sbf[0][0])[v] = bfeat[v];
    if (t < SH) swo[t] = Wo[t];
    for (int v = t; v < 256; v += 128) {
        int ii = v >> 4, jj = v & 15;
        int d = span_begin[i0 + ii] - span_begin[j0 + jj];
        d = d < 0 ? -d : d;
        int b;
        if (d < 5) b = d;
        else { b = (31 - __clz(d)) + 3; if (b > SNB - 1) b = SNB - 1; }
        sbid[ii][jj] = b;
        stile[v] = 0.0f;
    }
    __syncthreads();

    const int lane = t & 31;
    const int wave = t >> 5;        // 4 waves split the 64 h values
    const int hi   = lane >> 4;
    const int mn   = lane & 15;

    const uint4* urow = (const uint4*)&su_i[mn][0];
    const uint4* brow = (const uint4*)&su_j[mn][0];

    float sc[8];
#pragma unroll
    for (int e = 0; e < 8; ++e) sc[e] = 0.0f;

    for (int h = wave * 16; h < wave * 16 + 16; ++h) {
        const uint4* wrow = (const uint4*)(wpT + (size_t)h * SD);  // L2-resident
        __builtin_prefetch((const void*)(wpT + (size_t)(h + 1) * SD));
        FragC acc;
#pragma unroll
        for (int e = 0; e < 8; ++e) acc.f[e] = 0.0f;

#pragma unroll 4
        for (int kk = 0; kk < SD / 32; ++kk) {
            Frag16 a, b;
            int c0 = kk * 4 + hi;
            a.q[0] = pk_mul4(urow[c0],     wrow[c0]);       // A = u_i ⊙ Wp[:,h]  (bf16 packed)
            a.q[1] = pk_mul4(urow[c0 + 2], wrow[c0 + 2]);
            int bi = kk * 4 + hi;
            b.q[0] = brow[bi];
            b.q[1] = brow[bi + 2];
            acc.v = __builtin_amdgcn_wmma_f32_16x16x32_bf16(false, a.v, false, b.v,
                                                            (short)0, acc.v, false, false);
        }
        float wo_h = swo[h];
#pragma unroll
        for (int e = 0; e < 8; ++e) {   // C layout: M = hi*8+e, N = mn
            int m = hi * 8 + e;
            float vv = acc.f[e] + sleft[m][h] + sright[mn][h] + sbf[sbid[m][mn]][h];
            sc[e] += fmaxf(vv, 0.0f) * wo_h;
        }
    }
#pragma unroll
    for (int e = 0; e < 8; ++e) atomicAdd(&stile[(hi * 8 + e) * 16 + mn], sc[e]);  // ds_add_f32
    __syncthreads();

    float bo0 = bo[0];
    for (int v = t; v < 256; v += 128) {
        int ii = v >> 4, jj = v & 15;
        int gi = i0 + ii, gj = j0 + jj;
        float val = stile[v] + bo0 + svec[gi] + svec[gj];
        if (gi == gj) val = 0.0f;                  // cs * (1 - eye)
        cs[(size_t)gi * SK + gj] = val;
    }
}

// ---------------- masked dual softmax: p1 = softmax(cs - mask(j>=i)), p2 = softmax(cs - mask(j<=i))
__global__ __launch_bounds__(256)
void k_softmax(const float* __restrict__ cs, bf16_t* __restrict__ p1, bf16_t* __restrict__ p2) {
    __shared__ float red[256];
    const int i = blockIdx.x;
    const int t = threadIdx.x;
    const float* row = cs + (size_t)i * SK;
    for (int mode = 0; mode < 2; ++mode) {
        bf16_t* p = mode ? p2 : p1;
        int j0 = t, j1 = t + 256;
        bool k0 = mode ? (j0 > i) : (j0 < i);
        bool k1 = mode ? (j1 > i) : (j1 < i);
        float a = row[j0] - (k0 ? 0.0f : BIGF);
        float b = row[j1] - (k1 ? 0.0f : BIGF);
        red[t] = fmaxf(a, b); __syncthreads();
        for (int s = 128; s > 0; s >>= 1) { if (t < s) red[t] = fmaxf(red[t], red[t + s]); __syncthreads(); }
        float mx = red[0]; __syncthreads();
        float ea = __expf(a - mx), eb = __expf(b - mx);
        red[t] = ea + eb; __syncthreads();
        for (int s = 128; s > 0; s >>= 1) { if (t < s) red[t] += red[t + s]; __syncthreads(); }
        float inv = 1.0f / red[0]; __syncthreads();
        p[(size_t)i * SK + j0] = f2bf(ea * inv);
        p[(size_t)i * SK + j1] = f2bf(eb * inv);
    }
}

// ---------------- concat [u ; c] -> bf16 [SK][2*SD] ----------------
__global__ void k_concat(bf16_t* __restrict__ dst, const float* __restrict__ u,
                         const float* __restrict__ c) {
    int idx = blockIdx.x * blockDim.x + threadIdx.x;
    if (idx >= SK * 2 * SD) return;
    int i = idx / (2 * SD), col = idx - i * 2 * SD;
    float v = (col < SD) ? u[(size_t)i * SD + col] : c[(size_t)i * SD + (col - SD)];
    dst[idx] = f2bf(v);
}

// ---------------- gate combine: unew = g1*u+(1-g1)*c1 + g2*u+(1-g2)*c2 ----------------
__global__ void k_gate_combine(const float* __restrict__ u, const float* __restrict__ c1,
                               const float* __restrict__ c2, const float* __restrict__ G1,
                               const float* __restrict__ G2, float* __restrict__ unew,
                               bf16_t* __restrict__ u2bf) {
    int idx = blockIdx.x * blockDim.x + threadIdx.x;
    if (idx >= SK * SD) return;
    float uu = u[idx], a = c1[idx], b = c2[idx];
    float g1 = 1.0f / (1.0f + __expf(-G1[idx]));
    float g2 = 1.0f / (1.0f + __expf(-G2[idx]));
    float r = g1 * uu + (1.0f - g1) * a + g2 * uu + (1.0f - g2) * b;
    unew[idx] = r;
    u2bf[idx] = f2bf(r);
}

// ---------------- pruner score: pr[i] = u[i] @ Wpr + bpr ----------------
__global__ __launch_bounds__(512)
void k_rowdot(const float* __restrict__ u, const float* __restrict__ w,
              const float* __restrict__ bias, float* __restrict__ out) {
    int i = threadIdx.x;  // 512 rows, 1 block
    float s = bias[0];
    for (int d = 0; d < SD; ++d) s += u[(size_t)i * SD + d] * w[d];
    out[i] = s;
}

// ---------------- copy + masked scatter ----------------
__global__ void k_copy_f4(float4* __restrict__ dst, const float4* __restrict__ src, int n4) {
    int i = blockIdx.x * blockDim.x + threadIdx.x;
    if (i < n4) dst[i] = src[i];
}
__global__ void k_scatter(const int* __restrict__ prune, const int* __restrict__ span_len,
                          const float* __restrict__ upd, float* __restrict__ all_out) {
    int idx = blockIdx.x * blockDim.x + threadIdx.x;
    if (idx >= SK * SD) return;
    int k = idx / SD, d = idx - k * SD;
    if (k < span_len[0]) {
        int row = prune[k];
        all_out[(size_t)row * SD + d] = upd[idx];
    }
}

// ==================== launch ====================
extern "C" void kernel_launch(void* const* d_in, const int* in_sizes, int n_in,
                              void* d_out, int out_size, void* d_ws, size_t ws_size,
                              hipStream_t stream) {
    const float* all_span_vecs = (const float*)d_in[0];
    const float* span_vecs     = (const float*)d_in[1];
    const int*   span_begin    = (const int*)d_in[2];
    const float* span_scores   = (const float*)d_in[5];   // [K,1] -> [K]
    const int*   prune_indices = (const int*)d_in[6];
    const int*   span_lengths  = (const int*)d_in[7];
    const float* Wl = (const float*)d_in[9],  *bl = (const float*)d_in[10];
    const float* Wr = (const float*)d_in[11], *br = (const float*)d_in[12];
    const float* Wp = (const float*)d_in[13];
    const float* dist_emb = (const float*)d_in[14];
    const float* Wd = (const float*)d_in[15], *bd = (const float*)d_in[16];
    const float* Wo = (const float*)d_in[17], *bo = (const float*)d_in[18];
    const float* Wg1 = (const float*)d_in[19], *bg1 = (const float*)d_in[20];
    const float* Wg2 = (const float*)d_in[21], *bg2 = (const float*)d_in[22];
    const float* Wpr = (const float*)d_in[23], *bpr = (const float*)d_in[24];

    float* out_all = (float*)d_out;                        // [N][D]
    float* out_upd = out_all + (size_t)SN * SD;            // [K][D]
    float* out_cs  = out_upd + (size_t)SK * SD;            // [K][K]

    char* ws = (char*)d_ws; size_t off = 0;
    auto alloc = [&](size_t bytes) -> void* {
        void* p = ws + off; off = (off + bytes + 255) & ~(size_t)255; return p;
    };
    bf16_t* WlT  = (bf16_t*)alloc((size_t)SH * SD * 2);
    bf16_t* WrT  = (bf16_t*)alloc((size_t)SH * SD * 2);
    bf16_t* WpT  = (bf16_t*)alloc((size_t)SH * SD * 2);
    bf16_t* Wg1T = (bf16_t*)alloc((size_t)SD * 2 * SD * 2);
    bf16_t* Wg2T = (bf16_t*)alloc((size_t)SD * 2 * SD * 2);
    float*  bfeat = (float*)alloc((size_t)SNB * SH * 4);
    bf16_t* u_bf  = (bf16_t*)alloc((size_t)SK * SD * 2);
    bf16_t* uT_bf = (bf16_t*)alloc((size_t)SD * SK * 2);
    bf16_t* u2_bf = (bf16_t*)alloc((size_t)SK * SD * 2);
    float*  leftb  = (float*)alloc((size_t)SK * SH * 4);
    float*  rightb = (float*)alloc((size_t)SK * SH * 4);
    float*  left2  = (float*)alloc((size_t)SK * SH * 4);
    float*  right2 = (float*)alloc((size_t)SK * SH * 4);
    float*  cs1 = (float*)alloc((size_t)SK * SK * 4);
    bf16_t* p1  = (bf16_t*)alloc((size_t)SK * SK * 2);
    bf16_t* p2  = (bf16_t*)alloc((size_t)SK * SK * 2);
    float*  c1  = (float*)alloc((size_t)SK * SD * 4);
    float*  c2  = (float*)alloc((size_t)SK * SD * 4);
    bf16_t* A1  = (bf16_t*)alloc((size_t)SK * 2 * SD * 2);
    bf16_t* A2  = (bf16_t*)alloc((size_t)SK * 2 * SD * 2);
    float*  G1  = (float*)alloc((size_t)SK * SD * 4);
    float*  G2  = (float*)alloc((size_t)SK * SD * 4);
    float*  pr  = (float*)alloc((size_t)SK * 4);

    auto blks = [](int n) { return (n + 255) / 256; };

    // --- weight prep (bf16 + transpose for B^T GEMM operands) ---
    k_transpose_bf16<<<blks(SD * SH), 256, 0, stream>>>(WlT, Wl, SD, SH);
    k_transpose_bf16<<<blks(SD * SH), 256, 0, stream>>>(WrT, Wr, SD, SH);
    k_transpose_bf16<<<blks(SD * SH), 256, 0, stream>>>(WpT, Wp, SD, SH);
    k_transpose_bf16<<<blks(2 * SD * SD), 256, 0, stream>>>(Wg1T, Wg1, 2 * SD, SD);
    k_transpose_bf16<<<blks(2 * SD * SD), 256, 0, stream>>>(Wg2T, Wg2, 2 * SD, SD);
    k_bucket_feat<<<blks(SNB * SH), 256, 0, stream>>>(dist_emb, Wd, bd, bfeat);
    k_cvt_bf16<<<blks(SK * SD), 256, 0, stream>>>(u_bf, span_vecs, SK * SD);
    k_transpose_bf16<<<blks(SK * SD), 256, 0, stream>>>(uT_bf, span_vecs, SK, SD);

    // --- pass 1: projections, fused pair scorer, cs1 ---
    k_gemm_bt<<<dim3(SH / 16, SK / 16), 32, 0, stream>>>(u_bf, WlT, bl, leftb, SK, SH, SD);
    k_gemm_bt<<<dim3(SH / 16, SK / 16), 32, 0, stream>>>(u_bf, WrT, br, rightb, SK, SH, SD);
    k_scorer<<<dim3(SK / 16, SK / 16), 128, 0, stream>>>(u_bf, WpT, leftb, rightb, bfeat,
                                                         Wo, bo, span_scores, span_begin, cs1);

    // --- coref prop: softmax, attention matmuls, gates ---
    k_softmax<<<SK, 256, 0, stream>>>(cs1, p1, p2);
    k_gemm_bt<<<dim3(SD / 16, SK / 16), 32, 0, stream>>>(p1, uT_bf, nullptr, c1, SK, SD, SK);
    k_gemm_bt<<<dim3(SD / 16, SK / 16), 32, 0, stream>>>(p2, uT_bf, nullptr, c2, SK, SD, SK);
    k_concat<<<blks(SK * 2 * SD), 256, 0, stream>>>(A1, span_vecs, c1);
    k_concat<<<blks(SK * 2 * SD), 256, 0, stream>>>(A2, span_vecs, c2);
    k_gemm_bt<<<dim3(SD / 16, SK / 16), 32, 0, stream>>>(A1, Wg1T, bg1, G1, SK, SD, 2 * SD);
    k_gemm_bt<<<dim3(SD / 16, SK / 16), 32, 0, stream>>>(A2, Wg2T, bg2, G2, SK, SD, 2 * SD);
    k_gate_combine<<<blks(SK * SD), 256, 0, stream>>>(span_vecs, c1, c2, G1, G2, out_upd, u2_bf);

    // --- pass 2: rescore with updated vectors + pruner scores ---
    k_gemm_bt<<<dim3(SH / 16, SK / 16), 32, 0, stream>>>(u2_bf, WlT, bl, left2, SK, SH, SD);
    k_gemm_bt<<<dim3(SH / 16, SK / 16), 32, 0, stream>>>(u2_bf, WrT, br, right2, SK, SH, SD);
    k_rowdot<<<1, 512, 0, stream>>>(out_upd, Wpr, bpr, pr);
    k_scorer<<<dim3(SK / 16, SK / 16), 128, 0, stream>>>(u2_bf, WpT, left2, right2, bfeat,
                                                         Wo, bo, pr, span_begin, out_cs);

    // --- overwrite_spans: copy all spans, scatter updated rows ---
    k_copy_f4<<<blks(SN * SD / 4), 256, 0, stream>>>((float4*)out_all,
                                                     (const float4*)all_span_vecs, SN * SD / 4);
    k_scatter<<<blks(SK * SD), 256, 0, stream>>>(prune_indices, span_lengths, out_upd, out_all);
}
